// MultiHeadSelfAttention_10436770529900
// MI455X (gfx1250) — matrix operand order repaired and checked by reading, not compiled, not run on hardware
//
#include <hip/hip_runtime.h>
#include <math.h>

#define B_   2
#define S_   2048
#define D_   1024
#define H_   16
#define DK_  64
#define MTOT (B_ * S_)

typedef _Float16 v16h __attribute__((ext_vector_type(16)));
typedef _Float16 v8h  __attribute__((ext_vector_type(8)));
typedef float    v8f  __attribute__((ext_vector_type(8)));

// ---------------------------------------------------------------------------
// f32 -> f16 conversion
// ---------------------------------------------------------------------------
__global__ void cvt_f32_f16(const float* __restrict__ in, _Float16* __restrict__ out, int n) {
    int i = blockIdx.x * blockDim.x + threadIdx.x;
    if (i < n) out[i] = (_Float16)in[i];
}

// ---------------------------------------------------------------------------
// WMMA fragment loaders (layouts per CDNA5 ISA §7.12.2, wave32)
// A 16x32 f16: lane L: row m0+(L&15); halves [k0+8*(L>>4) .. +8) and [k0+16+8*(L>>4) .. +8)
// B 32x16 f16: lane L: column n0+(L&15) (== row of the transposed operand),
//              halves [k0+8*(L>>4) .. +8) and [k0+16+8*(L>>4) .. +8)
// ---------------------------------------------------------------------------
__device__ __forceinline__ v16h load_a_frag(const _Float16* __restrict__ A, int lda,
                                            int m0, int k0, int lane) {
    int row = m0 + (lane & 15);
    int kk  = k0 + ((lane >> 4) << 3);
    v8h lo = *(const v8h*)(A + (size_t)row * lda + kk);
    v8h hi = *(const v8h*)(A + (size_t)row * lda + kk + 16);
    return __builtin_shufflevector(lo, hi, 0,1,2,3,4,5,6,7,8,9,10,11,12,13,14,15);
}

__device__ __forceinline__ v16h load_b_frag(const _Float16* __restrict__ Wm, int ldb,
                                            int n0, int k0, int lane) {
    int col = n0 + (lane & 15);
    int kk  = k0 + ((lane >> 4) << 3);
    v8h lo = *(const v8h*)(Wm + (size_t)col * ldb + kk);
    v8h hi = *(const v8h*)(Wm + (size_t)col * ldb + kk + 16);
    return __builtin_shufflevector(lo, hi, 0,1,2,3,4,5,6,7,8,9,10,11,12,13,14,15);
}

__device__ __forceinline__ v8f wmma_f16(v16h a, v16h b, v8f c) {
    return __builtin_amdgcn_wmma_f32_16x16x32_f16(false, a, false, b, (short)0, c, false, false);
}

// ---------------------------------------------------------------------------
// GEMM: C[M,N] = (A[M,K] * W[N,K]^T + bias[N]) * oscale
// Block: 128 threads = 4 waves (2x2). Each wave computes 64x64 of C
// (16 accumulator tiles) -> 128x128 per block. 16 WMMA per 8 fragment
// loads per k-step (~32 flop / L2 byte). launch_bounds(…,1) lifts the
// occupancy-driven VGPR cap so the 128-VGPR accumulator set never spills.
// MODE 0: out = float [M,N]          (output projection -> d_out)
// MODE 1: out = f16  [b,h,s,dk]      (Q / K head-split layout)
// MODE 2: out = f16  [b,h,dk,s]      (V transposed, so P*V B-loads are contiguous)
// ---------------------------------------------------------------------------
template <int MODE>
__global__ __launch_bounds__(128, 1)
void gemm16(const _Float16* __restrict__ A, const _Float16* __restrict__ Wm,
            const float* __restrict__ bias, void* __restrict__ out,
            float oscale, int Mtot, int N, int K) {
    const int lane   = threadIdx.x & 31;
    const int wave   = threadIdx.x >> 5;
    const int m_base = blockIdx.x * 128 + (wave >> 1) * 64;
    const int n_base = blockIdx.y * 128 + (wave & 1)  * 64;

    v8f zero = {0.f,0.f,0.f,0.f,0.f,0.f,0.f,0.f};
    v8f acc[4][4];
#pragma unroll
    for (int i = 0; i < 4; ++i)
#pragma unroll
        for (int j = 0; j < 4; ++j) acc[i][j] = zero;

    for (int k0 = 0; k0 < K; k0 += 32) {
        v16h a[4];
#pragma unroll
        for (int i = 0; i < 4; ++i)
            a[i] = load_a_frag(A, K, m_base + i * 16, k0, lane);
#pragma unroll
        for (int j = 0; j < 4; ++j) {
            v16h b = load_b_frag(Wm, K, n_base + j * 16, k0, lane);
#pragma unroll
            for (int i = 0; i < 4; ++i)
                acc[i][j] = wmma_f16(a[i], b, acc[i][j]);
        }
    }

    const int hi = lane >> 4;
    const int ln = lane & 15;
#pragma unroll
    for (int i = 0; i < 4; ++i) {
#pragma unroll
        for (int j = 0; j < 4; ++j) {
            int n = n_base + j * 16 + ln;
            float bv = bias[n];
#pragma unroll
            for (int r = 0; r < 8; ++r) {
                int m = m_base + i * 16 + hi * 8 + r;
                float v = (acc[i][j][r] + bv) * oscale;
                if (MODE == 0) {
                    ((float*)out)[(size_t)m * N + n] = v;
                } else {
                    int b  = m >> 11;          // / S_
                    int s  = m & (S_ - 1);
                    int h  = n >> 6;           // / DK_
                    int dk = n & (DK_ - 1);
                    int bh = b * H_ + h;
                    _Float16* o = (_Float16*)out;
                    if (MODE == 1) o[((size_t)bh * S_ + s) * DK_ + dk] = (_Float16)v;
                    else           o[((size_t)bh * DK_ + dk) * S_ + s] = (_Float16)v;
                }
            }
        }
    }
}

// ---------------------------------------------------------------------------
// Flash-style attention: one wave = 16 query rows of one (b,h).
// Streams keys in chunks of 32 (low register pressure, no spills).
// 1/sqrt(Dk) is pre-folded into Q by the projection epilogue.
// Row max: one 4-step shuffle chain per row. Row SUM: WMMA against an
// all-ones B matrix (every lane of the C tile receives the row sum).
// P converts C-layout -> A-layout through a 1KB LDS tile.
// ---------------------------------------------------------------------------
__global__ __launch_bounds__(32, 1)
void attn16(const _Float16* __restrict__ Q, const _Float16* __restrict__ Km,
            const _Float16* __restrict__ Vt, _Float16* __restrict__ Ctx) {
    __shared__ __align__(16) _Float16 ptile[16 * 32];

    const int lane = threadIdx.x & 31;
    const int q0   = blockIdx.x * 16;
    const int bh   = blockIdx.y;
    const int b    = bh >> 4;   // / H_
    const int h    = bh & 15;

    const _Float16* Qb = Q  + (size_t)bh * S_ * DK_;
    const _Float16* Kb = Km + (size_t)bh * S_ * DK_;
    const _Float16* Vb = Vt + (size_t)bh * DK_ * S_;

    // Q fragments (reused across all key chunks): Dk=64 -> two k-steps
    v16h aq0 = load_a_frag(Qb, DK_, q0, 0,  lane);
    v16h aq1 = load_a_frag(Qb, DK_, q0, 32, lane);

    const v16h ones = {(_Float16)1.f,(_Float16)1.f,(_Float16)1.f,(_Float16)1.f,
                       (_Float16)1.f,(_Float16)1.f,(_Float16)1.f,(_Float16)1.f,
                       (_Float16)1.f,(_Float16)1.f,(_Float16)1.f,(_Float16)1.f,
                       (_Float16)1.f,(_Float16)1.f,(_Float16)1.f,(_Float16)1.f};

    v8f zero = {0.f,0.f,0.f,0.f,0.f,0.f,0.f,0.f};
    v8f cacc[4];
#pragma unroll
    for (int j = 0; j < 4; ++j) cacc[j] = zero;
    v8f lacc = zero;               // running row sums (all columns identical)

    float rm[8];
#pragma unroll
    for (int r = 0; r < 8; ++r) rm[r] = -1e30f;

    const int hi = lane >> 4;
    const int ln = lane & 15;

    for (int kc = 0; kc < S_; kc += 32) {
        // --- scores: 16 queries x 32 keys (2 tiles); Q already scaled ---
        v8f sc[2];
#pragma unroll
        for (int j = 0; j < 2; ++j) {
            v16h bk0 = load_b_frag(Kb, DK_, kc + j * 16, 0,  lane);
            v16h bk1 = load_b_frag(Kb, DK_, kc + j * 16, 32, lane);
            sc[j] = wmma_f16(aq0, bk0, zero);
            sc[j] = wmma_f16(aq1, bk1, sc[j]);
        }

        // --- online softmax (rows live within a 16-lane half-wave) ---
#pragma unroll
        for (int r = 0; r < 8; ++r) {
            float s0 = sc[0][r];
            float s1 = sc[1][r];
            float t = fmaxf(s0, s1);
#pragma unroll
            for (int off = 1; off < 16; off <<= 1)
                t = fmaxf(t, __shfl_xor(t, off, 32));
            float newm = fmaxf(rm[r], t);
            float corr = __expf(rm[r] - newm);
            rm[r] = newm;
            lacc[r] *= corr;
#pragma unroll
            for (int j = 0; j < 4; ++j) cacc[j][r] *= corr;

            int rowoff = (hi * 8 + r) * 32 + ln;
            ptile[rowoff]      = (_Float16)__expf(s0 - newm);
            ptile[rowoff + 16] = (_Float16)__expf(s1 - newm);
        }
        __syncthreads();

        // re-read P in A-layout (one 16x32 fragment)
        v16h pa;
        {
            int prow = ln;
            int pk   = hi * 8;
            v8h lo = *(const v8h*)(ptile + prow * 32 + pk);
            v8h hh = *(const v8h*)(ptile + prow * 32 + 16 + pk);
            pa = __builtin_shufflevector(lo, hh, 0,1,2,3,4,5,6,7,8,9,10,11,12,13,14,15);
        }
        __syncthreads();

        // row sums via WMMA with all-ones B (every lane gets the row sum)
        lacc = wmma_f16(pa, ones, lacc);

        // ctx += P * V  (V stored transposed [dk, s] -> contiguous B loads)
#pragma unroll
        for (int j = 0; j < 4; ++j) {
            v16h bv = load_b_frag(Vb, S_, j * 16, kc, lane);
            cacc[j] = wmma_f16(pa, bv, cacc[j]);
        }
    }

    // --- normalize and write ctx as f16 [B,S,D] for the output projection ---
    float inv[8];
#pragma unroll
    for (int r = 0; r < 8; ++r) inv[r] = 1.f / lacc[r];
#pragma unroll
    for (int j = 0; j < 4; ++j) {
        int dv = j * 16 + ln;
#pragma unroll
        for (int r = 0; r < 8; ++r) {
            float v = cacc[j][r] * inv[r];
            int s = q0 + hi * 8 + r;
            Ctx[((size_t)(b * S_ + s)) * D_ + h * DK_ + dv] = (_Float16)v;
        }
    }
}

// ---------------------------------------------------------------------------
// launch
// ---------------------------------------------------------------------------
extern "C" void kernel_launch(void* const* d_in, const int* in_sizes, int n_in,
                              void* d_out, int out_size, void* d_ws, size_t ws_size,
                              hipStream_t stream) {
    const float* x  = (const float*)d_in[0];
    const float* Wq = (const float*)d_in[1];
    const float* bq = (const float*)d_in[2];
    const float* Wk = (const float*)d_in[3];
    const float* bk = (const float*)d_in[4];
    const float* Wv = (const float*)d_in[5];
    const float* bv = (const float*)d_in[6];
    const float* Wo = (const float*)d_in[7];
    const float* bo = (const float*)d_in[8];

    const size_t NX = (size_t)MTOT * D_;   // 4,194,304
    const size_t NW = (size_t)D_ * D_;     // 1,048,576

    _Float16* ws   = (_Float16*)d_ws;
    _Float16* xh   = ws;                 // NX
    _Float16* wqh  = xh  + NX;           // NW
    _Float16* wkh  = wqh + NW;
    _Float16* wvh  = wkh + NW;
    _Float16* woh  = wvh + NW;
    _Float16* Qh   = woh + NW;           // NX  [b,h,s,dk]  (pre-scaled by 1/8)
    _Float16* Kh   = Qh  + NX;           // NX  [b,h,s,dk]
    _Float16* Vth  = Kh  + NX;           // NX  [b,h,dk,s]
    _Float16* Ctxh = Vth + NX;           // NX  [b,s,d]

    // f32 -> f16 staging
    cvt_f32_f16<<<(int)((NX + 255) / 256), 256, 0, stream>>>(x,  xh,  (int)NX);
    cvt_f32_f16<<<(int)((NW + 255) / 256), 256, 0, stream>>>(Wq, wqh, (int)NW);
    cvt_f32_f16<<<(int)((NW + 255) / 256), 256, 0, stream>>>(Wk, wkh, (int)NW);
    cvt_f32_f16<<<(int)((NW + 255) / 256), 256, 0, stream>>>(Wv, wvh, (int)NW);
    cvt_f32_f16<<<(int)((NW + 255) / 256), 256, 0, stream>>>(Wo, woh, (int)NW);

    dim3 gblock(128);
    dim3 ggrid(MTOT / 128, D_ / 128);  // 32 x 8

    // Q carries the 1/sqrt(Dk) softmax scale
    gemm16<1><<<ggrid, gblock, 0, stream>>>(xh, wqh, bq, Qh,  0.125f, MTOT, D_, D_);
    gemm16<1><<<ggrid, gblock, 0, stream>>>(xh, wkh, bk, Kh,  1.0f,   MTOT, D_, D_);
    gemm16<2><<<ggrid, gblock, 0, stream>>>(xh, wvh, bv, Vth, 1.0f,   MTOT, D_, D_);

    attn16<<<dim3(S_ / 16, B_ * H_), 32, 0, stream>>>(Qh, Kh, Vth, Ctxh);

    gemm16<0><<<ggrid, gblock, 0, stream>>>(Ctxh, woh, bo, d_out, 1.0f, MTOT, D_, D_);
}
